// SDPAttention_1529008357925
// MI455X (gfx1250) — compile-verified, matrix operand order checked by reading, not run on hardware
//
#include <hip/hip_runtime.h>
#include <hip/hip_bf16.h>

typedef __attribute__((ext_vector_type(16))) _Float16 v16h;
typedef __attribute__((ext_vector_type(8)))  _Float16 v8h;
typedef __attribute__((ext_vector_type(8)))  float    v8f;

#define WMMA_F16(a, b, c) \
  __builtin_amdgcn_wmma_f32_16x16x32_f16(false, (a), false, (b), (short)0, (c), false, false)

constexpr int Bc = 8, Hh = 16, Lc = 1024, Ec = 1024, Ac = 64, E3 = 3072;

// ---- contiguous fragment loader (ISA 7.12.2 16-bit A/B packing) ----
// lane%16 = row (A) or col (B); K base = (lane/16)*8; elements K = base+{0..7,16..23}
__device__ __forceinline__ v16h frag_row_f16(const _Float16* __restrict__ p0, int ld,
                                             int row, int k0, int lane) {
  const _Float16* p = p0 + (size_t)row * ld + k0 + ((lane >> 4) << 3);
  v16h f;
#pragma unroll
  for (int j = 0; j < 8; ++j) f[j] = p[j];
#pragma unroll
  for (int j = 0; j < 8; ++j) f[j + 8] = p[16 + j];
  return f;
}

// ---- fp32 -> f16 conversion (vectorized x4) ----
__global__ void cvt_f32_f16(const float* __restrict__ s, _Float16* __restrict__ d, int n) {
  int i = (blockIdx.x * blockDim.x + threadIdx.x) * 4;
  if (i < n) {
    float4 v = *(const float4*)(s + i);
    d[i + 0] = (_Float16)v.x;
    d[i + 1] = (_Float16)v.y;
    d[i + 2] = (_Float16)v.z;
    d[i + 3] = (_Float16)v.w;
  }
}

// ---- fp32 (K x N) -> f16 transposed (N x K), LDS-tiled ----
__global__ __launch_bounds__(256) void cvt_transpose_f16(const float* __restrict__ src,
                                                         _Float16* __restrict__ dst,
                                                         int K, int N) {
  __shared__ _Float16 tile[16][17];           // +1 pad: no bank conflicts
  int kb = blockIdx.y * 16, nb = blockIdx.x * 16;
  int tx = threadIdx.x & 15, ty = threadIdx.x >> 4;
  tile[ty][tx] = (_Float16)src[(size_t)(kb + ty) * N + (nb + tx)];
  __syncthreads();
  dst[(size_t)(nb + ty) * K + (kb + tx)] = tile[tx][ty];
}

// ---- QKV projection: X(8192x1024) @ Wt(3072x1024), 64x64 tile per wave ----
// N-group (64 cols, 64-aligned) lies in exactly one q/k/v part of one head;
// M-group (64 rows, 64-aligned) lies in exactly one batch element.
__global__ __launch_bounds__(256) void qkv_gemm(const _Float16* __restrict__ X,
                                                const _Float16* __restrict__ Wt,
                                                _Float16* __restrict__ Q,
                                                _Float16* __restrict__ Kd,
                                                _Float16* __restrict__ Vt) {
  int lane = threadIdx.x & 31;
  int wid  = (blockIdx.x << 3) + (threadIdx.x >> 5);
  int ng = wid % (E3 / 64);                    // 48 N-groups
  int mg = wid / (E3 / 64);                    // 128 M-groups
  int col = lane & 15;
  int rb  = (lane >> 4) << 3;

  v8f acc[4][4];
#pragma unroll
  for (int mt = 0; mt < 4; ++mt)
#pragma unroll
    for (int nt = 0; nt < 4; ++nt) acc[mt][nt] = (v8f){};

#pragma unroll 1
  for (int kk = 0; kk < Ec; kk += 32) {
    v16h af[4], bf[4];
#pragma unroll
    for (int mt = 0; mt < 4; ++mt)
      af[mt] = frag_row_f16(X, Ec, mg * 64 + mt * 16 + col, kk, lane);
#pragma unroll
    for (int nt = 0; nt < 4; ++nt)
      bf[nt] = frag_row_f16(Wt, Ec, ng * 64 + nt * 16 + col, kk, lane);
#pragma unroll
    for (int mt = 0; mt < 4; ++mt)
#pragma unroll
      for (int nt = 0; nt < 4; ++nt)
        acc[mt][nt] = WMMA_F16(af[mt], bf[nt], acc[mt][nt]);
  }

  int c0   = ng * 64;
  int h    = c0 / (3 * Ac);
  int part = (c0 % (3 * Ac)) >> 6;             // uniform for the whole wave
  int bb   = (mg * 64) >> 10;                  // uniform
  int l00  = (mg * 64) & 1023;

  if (part == 2) {
    // V transposed per head: [b][h][a][l]; 8 consecutive l per lane -> b128 store
#pragma unroll
    for (int mt = 0; mt < 4; ++mt) {
      int l0 = l00 + mt * 16 + rb;
#pragma unroll
      for (int nt = 0; nt < 4; ++nt) {
        int aa = nt * 16 + col;
        v8h pack;
#pragma unroll
        for (int r = 0; r < 8; ++r) pack[r] = (_Float16)acc[mt][nt][r];
        *(v8h*)(Vt + (((size_t)bb * Hh + h) * Ac + aa) * Lc + l0) = pack;
      }
    }
  } else {
    _Float16* dst = (part == 0) ? Q : Kd;
#pragma unroll
    for (int mt = 0; mt < 4; ++mt)
#pragma unroll
      for (int nt = 0; nt < 4; ++nt) {
        int aa = nt * 16 + col;
#pragma unroll
        for (int r = 0; r < 8; ++r) {
          int l = l00 + mt * 16 + rb + r;
          dst[(((size_t)bb * Hh + h) * Lc + l) * Ac + aa] = (_Float16)acc[mt][nt][r];
        }
      }
  }
}

// ---- fused flash attention: one wave = (b, h, 16-query tile) over all 1024 keys ----
__global__ __launch_bounds__(256) void attn_fused(const _Float16* __restrict__ Q,
                                                  const _Float16* __restrict__ Kd,
                                                  const _Float16* __restrict__ Vt,
                                                  const float* __restrict__ bias,
                                                  const int* __restrict__ mask,
                                                  _Float16* __restrict__ ctx) {
  __shared__ float psh[8 * 16 * 32];
  int lane  = threadIdx.x & 31;
  int wslot = threadIdx.x >> 5;
  int wid = (blockIdx.x << 3) + wslot;
  int b   = wid >> 10;
  int rem = wid & 1023;
  int h  = rem >> 6;
  int qt = rem & 63;

  const _Float16* Qh  = Q  + ((size_t)b * Hh + h) * Lc * Ac;
  const _Float16* Kh  = Kd + ((size_t)b * Hh + h) * Lc * Ac;
  const _Float16* Vth = Vt + ((size_t)b * Hh + h) * Ac * Lc;   // [a][l]
  const float* biash = bias + ((size_t)b * Hh + h) * Lc * Lc;
  const int* maskb   = mask + (size_t)b * Lc * Lc;

  int col = lane & 15;
  int rb  = (lane >> 4) << 3;
  v16h q0 = frag_row_f16(Qh, Ac, qt * 16 + col, 0, lane);
  v16h q1 = frag_row_f16(Qh, Ac, qt * 16 + col, 32, lane);

  v8f acc0 = {}, acc1 = {}, acc2 = {}, acc3 = {};
  float mrow[8], lrow[8];
#pragma unroll
  for (int r = 0; r < 8; ++r) { mrow[r] = -3.0e38f; lrow[r] = 0.f; }

  float* sp = psh + wslot * (16 * 32);

  for (int kt = 0; kt < Lc; kt += 32) {
    // ---- logits: two 16-key tiles; K^T B-frags contiguous along a-dim ----
    v8f lg[2];
#pragma unroll
    for (int t = 0; t < 2; ++t) {
      int key0 = kt + t * 16;
      v16h k0f = frag_row_f16(Kh, Ac, key0 + col, 0, lane);
      v16h k1f = frag_row_f16(Kh, Ac, key0 + col, 32, lane);
      v8f z = {};
      z = WMMA_F16(q0, k0f, z);
      z = WMMA_F16(q1, k1f, z);
#pragma unroll
      for (int r = 0; r < 8; ++r) {
        int qrow = qt * 16 + rb + r;
        int key  = key0 + col;
        float bv = biash[(size_t)qrow * Lc + key];
        int   mk = maskb[(size_t)qrow * Lc + key];
        float val = z[r] * 0.125f + bv;   // SCALE = A^-0.5 = 0.125
        z[r] = mk ? -3.0e38f : val;       // mask==true -> finfo.min
      }
      lg[t] = z;
    }
    // ---- online softmax row stats (reduce across 16 lanes of each half) ----
    float tmax[8];
#pragma unroll
    for (int r = 0; r < 8; ++r) tmax[r] = fmaxf(lg[0][r], lg[1][r]);
#pragma unroll
    for (int msk = 1; msk < 16; msk <<= 1)
#pragma unroll
      for (int r = 0; r < 8; ++r) tmax[r] = fmaxf(tmax[r], __shfl_xor(tmax[r], msk, 32));

    float alpha[8], newm[8];
#pragma unroll
    for (int r = 0; r < 8; ++r) {
      newm[r]  = fmaxf(mrow[r], tmax[r]);
      alpha[r] = __expf(mrow[r] - newm[r]);
      mrow[r]  = newm[r];
    }
    float p0[8], p1[8], srow[8];
#pragma unroll
    for (int r = 0; r < 8; ++r) {
      p0[r] = __expf(lg[0][r] - newm[r]);
      p1[r] = __expf(lg[1][r] - newm[r]);
      srow[r] = p0[r] + p1[r];
    }
#pragma unroll
    for (int msk = 1; msk < 16; msk <<= 1)
#pragma unroll
      for (int r = 0; r < 8; ++r) srow[r] += __shfl_xor(srow[r], msk, 32);
#pragma unroll
    for (int r = 0; r < 8; ++r) lrow[r] = lrow[r] * alpha[r] + srow[r];
#pragma unroll
    for (int r = 0; r < 8; ++r) {
      acc0[r] *= alpha[r]; acc1[r] *= alpha[r];
      acc2[r] *= alpha[r]; acc3[r] *= alpha[r];
    }
    // ---- transpose probs C-layout -> A-fragment through LDS ----
#pragma unroll
    for (int r = 0; r < 8; ++r) {
      sp[(rb + r) * 32 + col]      = p0[r];
      sp[(rb + r) * 32 + 16 + col] = p1[r];
    }
    asm volatile("s_wait_dscnt 0" ::: "memory");
    v16h pf;
    {
      const float* pr = sp + (size_t)(lane & 15) * 32 + rb;
#pragma unroll
      for (int j = 0; j < 8; ++j) pf[j]     = (_Float16)pr[j];
#pragma unroll
      for (int j = 0; j < 8; ++j) pf[j + 8] = (_Float16)pr[16 + j];
    }
    // ---- P(16x32) @ V(32x64): Vt[a][l] makes B-frags contiguous along key ----
#pragma unroll
    for (int t = 0; t < 4; ++t) {
      v16h vf = frag_row_f16(Vth, Lc, t * 16 + col, kt, lane);
      if      (t == 0) acc0 = WMMA_F16(pf, vf, acc0);
      else if (t == 1) acc1 = WMMA_F16(pf, vf, acc1);
      else if (t == 2) acc2 = WMMA_F16(pf, vf, acc2);
      else             acc3 = WMMA_F16(pf, vf, acc3);
    }
    if (kt + 32 < Lc)   // stream next bias tile into cache ahead of use
      __builtin_prefetch(&biash[(size_t)(qt * 16 + rb) * Lc + kt + 32], 0, 0);
  }
  // ---- normalize, store ctx f16 [b][l][h*64+a] ----
#pragma unroll
  for (int r = 0; r < 8; ++r) {
    int qrow = qt * 16 + rb + r;
    float inv = 1.0f / lrow[r];
    size_t base = ((size_t)b * Lc + qrow) * Ec + (size_t)h * Ac;
    ctx[base +  0 + col] = (_Float16)(acc0[r] * inv);
    ctx[base + 16 + col] = (_Float16)(acc1[r] * inv);
    ctx[base + 32 + col] = (_Float16)(acc2[r] * inv);
    ctx[base + 48 + col] = (_Float16)(acc3[r] * inv);
  }
}

// ---- output projection: ctx(8192x1024) @ WoutT(1024x1024), 64x64 per wave ----
__global__ __launch_bounds__(256) void out_gemm(const _Float16* __restrict__ X,
                                                const _Float16* __restrict__ Wt,
                                                float* __restrict__ out) {
  int lane = threadIdx.x & 31;
  int wid  = (blockIdx.x << 3) + (threadIdx.x >> 5);
  int ng = wid & 15;                           // 16 N-groups
  int mg = wid >> 4;                           // 128 M-groups
  int col = lane & 15;
  int rb  = (lane >> 4) << 3;

  v8f acc[4][4];
#pragma unroll
  for (int mt = 0; mt < 4; ++mt)
#pragma unroll
    for (int nt = 0; nt < 4; ++nt) acc[mt][nt] = (v8f){};

#pragma unroll 1
  for (int kk = 0; kk < Ec; kk += 32) {
    v16h af[4], bf[4];
#pragma unroll
    for (int mt = 0; mt < 4; ++mt)
      af[mt] = frag_row_f16(X, Ec, mg * 64 + mt * 16 + col, kk, lane);
#pragma unroll
    for (int nt = 0; nt < 4; ++nt)
      bf[nt] = frag_row_f16(Wt, Ec, ng * 64 + nt * 16 + col, kk, lane);
#pragma unroll
    for (int mt = 0; mt < 4; ++mt)
#pragma unroll
      for (int nt = 0; nt < 4; ++nt)
        acc[mt][nt] = WMMA_F16(af[mt], bf[nt], acc[mt][nt]);
  }
#pragma unroll
  for (int mt = 0; mt < 4; ++mt)
#pragma unroll
    for (int nt = 0; nt < 4; ++nt)
#pragma unroll
      for (int r = 0; r < 8; ++r) {
        int m = mg * 64 + mt * 16 + rb + r;
        out[(size_t)m * Ec + ng * 64 + nt * 16 + col] = acc[mt][nt][r];
      }
}

extern "C" void kernel_launch(void* const* d_in, const int* in_sizes, int n_in,
                              void* d_out, int out_size, void* d_ws, size_t ws_size,
                              hipStream_t stream) {
  const float* emb  = (const float*)d_in[0];
  const int*   mask = (const int*)  d_in[1];
  const float* bias = (const float*)d_in[2];
  const float* Wqkv = (const float*)d_in[3];
  const float* Wout = (const float*)d_in[4];
  float* out = (float*)d_out;

  char* ws = (char*)d_ws;
  size_t off = 0;
  auto alloc = [&](size_t bytes) -> void* {
    void* p = ws + off;
    off = (off + bytes + 255) & ~(size_t)255;
    return p;
  };
  const size_t NTOK = (size_t)Bc * Lc;            // 8192
  _Float16* emb16 = (_Float16*)alloc(NTOK * Ec * 2);
  _Float16* WqT   = (_Float16*)alloc((size_t)E3 * Ec * 2);  // (3072 x 1024)
  _Float16* WoT   = (_Float16*)alloc((size_t)Ec * Ec * 2);  // (1024 x 1024)
  _Float16* Qb    = (_Float16*)alloc((size_t)Bc * Hh * Lc * Ac * 2);
  _Float16* Kb    = (_Float16*)alloc((size_t)Bc * Hh * Lc * Ac * 2);
  _Float16* Vt    = (_Float16*)alloc((size_t)Bc * Hh * Ac * Lc * 2);
  _Float16* ctx16 = (_Float16*)alloc(NTOK * Ec * 2);

  // fp32 -> f16 staging (weights transposed so all B-frags are contiguous)
  {
    int n = (int)(NTOK * Ec);
    cvt_f32_f16<<<(n / 4 + 255) / 256, 256, 0, stream>>>(emb, emb16, n);
  }
  cvt_transpose_f16<<<dim3(E3 / 16, Ec / 16), 256, 0, stream>>>(Wqkv, WqT, Ec, E3);
  cvt_transpose_f16<<<dim3(Ec / 16, Ec / 16), 256, 0, stream>>>(Wout, WoT, Ec, Ec);

  // QKV projection: 128 M-groups x 48 N-groups = 6144 waves / 8 per block
  qkv_gemm<<<(128 * 48) / 8, 256, 0, stream>>>(emb16, WqT, Qb, Kb, Vt);

  // fused attention: 8 * 16 * 64 = 8192 waves / 8 per block
  attn_fused<<<8192 / 8, 256, 0, stream>>>(Qb, Kb, Vt, bias, mask, ctx16);

  // output projection: 128 x 16 = 2048 waves / 8 per block
  out_gemm<<<(128 * 16) / 8, 256, 0, stream>>>(ctx16, WoT, out);
}
